// EpiEPMP_83983790506437
// MI455X (gfx1250) — matrix operand order repaired and checked by reading, not compiled, not run on hardware
//
#include <hip/hip_runtime.h>

#define N1 20000
#define N2 20000
#define NN 40000
#define FDIM 256
#define E1C 320000
#define E2C 320000
#define EDC 640000
#define EEC (EDC + NN)
#define EPSV 1e-5f

typedef _Float16 half_t;
typedef half_t v16h __attribute__((ext_vector_type(16)));
typedef float v8f __attribute__((ext_vector_type(8)));

// ---------- helpers ----------
__device__ __forceinline__ int flip_f(float f) {
  int i = __float_as_int(f);
  return (i >= 0) ? i : (i ^ 0x7FFFFFFF);
}
__device__ __forceinline__ float unflip_f(int i) {
  int j = (i >= 0) ? i : (i ^ 0x7FFFFFFF);
  return __int_as_float(j);
}

// ---------- GEMM operand prep: f32 -> f16 in WMMA fragment order ----------
// A fragment layout: Ah[((mt*8 + kc)*32 + lane)*16 + e]
//   lane 0-15: row = mt*16+lane,    K = kc*32 + e      (e = 0..15)
//   lane16-31: row = mt*16+lane-16, K = kc*32 + 16 + e
__global__ void k_prep_a(const float* __restrict__ A, half_t* __restrict__ Ah, int M) {
  int t = blockIdx.x * 256 + threadIdx.x;  // total = (M/16)*8*32 = M*16
  if (t >= M * 16) return;
  int lane = t & 31;
  int kc = (t >> 5) & 7;
  int mt = t >> 8;
  int row = mt * 16 + (lane & 15);
  int kbase = kc * 32 + ((lane & 16) ? 16 : 0);
  const float* ap = A + (size_t)row * FDIM + kbase;
  half_t* o = Ah + (size_t)t * 16;
#pragma unroll
  for (int e = 0; e < 8; ++e) {
    o[e] = (half_t)ap[e];
    o[8 + e] = (half_t)ap[8 + e];
  }
}
// B fragment layout: Wh[((nt*8 + kc)*32 + lane)*16 + e]
//   lane 0-15: col = nt*16+lane,    K = kc*32 + e      (e = 0..15)
//   lane16-31: col = nt*16+lane-16, K = kc*32 + 16 + e
__global__ void k_prep_w(const float* __restrict__ W, half_t* __restrict__ Wh) {
  int t = blockIdx.x * 256 + threadIdx.x;  // total = 16*8*32 = 4096
  if (t >= 4096) return;
  int lane = t & 31;
  int kc = (t >> 5) & 7;
  int nt = t >> 8;
  int col = nt * 16 + (lane & 15);
  int kbase = kc * 32 + ((lane & 16) ? 16 : 0);
  half_t* o = Wh + (size_t)t * 16;
#pragma unroll
  for (int e = 0; e < 16; ++e) o[e] = (half_t)W[(size_t)(kbase + e) * FDIM + col];
}

// ---------- GEMM: C[M x 256] = A @ W, wave computes 16x64 via 4 accumulators ----------
__global__ __launch_bounds__(256) void k_gemm_wmma(const half_t* __restrict__ Ah,
                                                   const half_t* __restrict__ Wh,
                                                   float* __restrict__ C, int M) {
  int lane = threadIdx.x & 31;
  int wv = blockIdx.x * 8 + (threadIdx.x >> 5);
  int mt = wv >> 2;  // 4 column-groups (of 4 N-tiles) per M strip
  int g = wv & 3;
  if (mt * 16 >= M) return;  // wave-uniform
  v8f acc[4] = {};
  const half_t* ab = Ah + (size_t)mt * (8 * 32 * 16);
  const half_t* wb = Wh + (size_t)g * (4 * 8 * 32 * 16);
#pragma unroll 2
  for (int kc = 0; kc < 8; ++kc) {
    // issue all operand loads for this k-step up front so they form one clause
    // and later WMMAs overlap the tail of the loads
    v16h a = *(const v16h*)(ab + ((size_t)kc * 32 + lane) * 16);
    v16h b0 = *(const v16h*)(wb + (((size_t)0 * 8 + kc) * 32 + lane) * 16);
    v16h b1 = *(const v16h*)(wb + (((size_t)1 * 8 + kc) * 32 + lane) * 16);
    v16h b2 = *(const v16h*)(wb + (((size_t)2 * 8 + kc) * 32 + lane) * 16);
    v16h b3 = *(const v16h*)(wb + (((size_t)3 * 8 + kc) * 32 + lane) * 16);
    acc[0] = __builtin_amdgcn_wmma_f32_16x16x32_f16(false, a, false, b0, (short)0, acc[0],
                                                    false, false);
    acc[1] = __builtin_amdgcn_wmma_f32_16x16x32_f16(false, a, false, b1, (short)0, acc[1],
                                                    false, false);
    acc[2] = __builtin_amdgcn_wmma_f32_16x16x32_f16(false, a, false, b2, (short)0, acc[2],
                                                    false, false);
    acc[3] = __builtin_amdgcn_wmma_f32_16x16x32_f16(false, a, false, b3, (short)0, acc[3],
                                                    false, false);
  }
  int rbase = mt * 16 + ((lane & 16) ? 8 : 0);
  int cl = lane & 15;
#pragma unroll
  for (int q = 0; q < 4; ++q) {
    int cn = (g * 4 + q) * 16 + cl;
#pragma unroll
    for (int i = 0; i < 8; ++i) C[(size_t)(rbase + i) * FDIM + cn] = acc[q][i];
  }
}

// ---------- elementwise / fills ----------
__global__ void k_fill_f(float* p, float v, int n) {
  int i = blockIdx.x * 256 + threadIdx.x;
  if (i < n) p[i] = v;
}
__global__ void k_fill_i(int* p, int v, int n) {
  int i = blockIdx.x * 256 + threadIdx.x;
  if (i < n) p[i] = v;
}
__global__ void k_relu_ip(float* x, int n) {
  int i = blockIdx.x * 256 + threadIdx.x;
  if (i < n) x[i] = fmaxf(x[i], 0.f);
}

// ---------- GCN ----------
__global__ void k_degree(const int* __restrict__ dst, int E, float* __restrict__ deg) {
  int i = blockIdx.x * 256 + threadIdx.x;
  if (i < E) atomicAdd(&deg[dst[i]], 1.0f);
}
__global__ void k_rsqrt_ip(float* d, int n) {
  int i = blockIdx.x * 256 + threadIdx.x;
  if (i < n) d[i] = rsqrtf(fmaxf(d[i], 1.0f));
}
// out = h * dinv^2 (self loop) + bias
__global__ void k_gcn_init(const float* __restrict__ h, const float* __restrict__ dinv,
                           const float* __restrict__ bias, float* __restrict__ out,
                           int rows) {
  int i = blockIdx.x * 256 + threadIdx.x;
  if (i >= rows * FDIM) return;
  int r = i >> 8, f = i & 255;
  float w = dinv[r];
  out[i] = h[i] * w * w + bias[f];
}
// wave per edge: out[dst] += h[src] * dinv[src]*dinv[dst]
__global__ __launch_bounds__(256) void k_gcn_agg(const int* __restrict__ src,
                                                 const int* __restrict__ dst, int E,
                                                 const float* __restrict__ h,
                                                 const float* __restrict__ dinv,
                                                 float* __restrict__ out) {
  int lane = threadIdx.x & 31;
  int e = blockIdx.x * 8 + (threadIdx.x >> 5);
  if (e >= E) return;
  int s = src[e], d = dst[e];
  float w = dinv[s] * dinv[d];
  const float* hp = h + (size_t)s * FDIM;
  float* op = out + (size_t)d * FDIM;
#pragma unroll
  for (int j = 0; j < 8; ++j) {
    int f = lane + 32 * j;
    atomicAdd(&op[f], hp[f] * w);
  }
}

// ---------- BatchNorm (over rows, per column) ----------
__global__ void k_colstats(const float* __restrict__ x, int rows, float* __restrict__ sum,
                           float* __restrict__ sq) {
  int c = threadIdx.x;  // blockDim = 256
  int r0 = blockIdx.x * 64;
  int rend = r0 + 64;
  if (rend > rows) rend = rows;
  float s = 0.f, s2 = 0.f;
  for (int r = r0; r < rend; ++r) {
    float v = x[(size_t)r * FDIM + c];
    s += v;
    s2 += v * v;
  }
  atomicAdd(&sum[c], s);
  atomicAdd(&sq[c], s2);
}
// stats over virtual concat([xa, xb], axis=1): 512 columns
__global__ void k_colstats2(const float* __restrict__ xa, const float* __restrict__ xb,
                            int rows, float* __restrict__ sum, float* __restrict__ sq) {
  int c = threadIdx.x;  // blockDim = 512
  int r0 = blockIdx.x * 64;
  int rend = r0 + 64;
  if (rend > rows) rend = rows;
  float s = 0.f, s2 = 0.f;
  for (int r = r0; r < rend; ++r) {
    float v = (c < FDIM) ? xa[(size_t)r * FDIM + c] : xb[(size_t)r * FDIM + (c - FDIM)];
    s += v;
    s2 += v * v;
  }
  atomicAdd(&sum[c], s);
  atomicAdd(&sq[c], s2);
}
__global__ void k_bn_finalize(const float* __restrict__ sum, const float* __restrict__ sq,
                              int rows, int C, float* __restrict__ mean,
                              float* __restrict__ rstd) {
  int i = blockIdx.x * 256 + threadIdx.x;
  if (i >= C) return;
  float inv = 1.0f / (float)rows;
  float m = sum[i] * inv;
  float var = sq[i] * inv - m * m;
  mean[i] = m;
  rstd[i] = rsqrtf(var + EPSV);
}
__global__ void k_bn_apply_relu(float* __restrict__ x, const float* __restrict__ mean,
                                const float* __restrict__ rstd,
                                const float* __restrict__ gamma,
                                const float* __restrict__ beta, int rows) {
  int i = blockIdx.x * 256 + threadIdx.x;
  if (i >= rows * FDIM) return;
  int f = i & 255;
  float v = (x[i] - mean[f]) * rstd[f] * gamma[f] + beta[f];
  x[i] = fmaxf(v, 0.f);
}

// ---------- GAT ----------
// wave per row: o1[r] = h[r,:] . a1 ; o2[r] = h[r,:] . a2
__global__ __launch_bounds__(256) void k_rowdot2(const float* __restrict__ h,
                                                 const float* __restrict__ a1,
                                                 const float* __restrict__ a2,
                                                 float* __restrict__ o1,
                                                 float* __restrict__ o2, int rows) {
  int lane = threadIdx.x & 31;
  int r = blockIdx.x * 8 + (threadIdx.x >> 5);
  if (r >= rows) return;
  const float* hp = h + (size_t)r * FDIM;
  float s1 = 0.f, s2 = 0.f;
#pragma unroll
  for (int j = 0; j < 8; ++j) {
    int f = lane + 32 * j;
    float v = hp[f];
    s1 += v * a1[f];
    s2 += v * a2[f];
  }
  for (int off = 16; off > 0; off >>= 1) {
    s1 += __shfl_xor(s1, off, 32);
    s2 += __shfl_xor(s2, off, 32);
  }
  if (lane == 0) {
    o1[r] = s1;
    o2[r] = s2;
  }
}
// e = leaky_relu(hs[src]+hd[dst]); segment max via int-key atomicMax
__global__ void k_edge_logits(const int* __restrict__ ei, int E, int n,
                              const float* __restrict__ hs, const float* __restrict__ hd,
                              float* __restrict__ ev, int* __restrict__ mkey) {
  int i = blockIdx.x * 256 + threadIdx.x;
  if (i >= E + n) return;
  int s, d;
  if (i < E) {
    s = ei[i];
    d = ei[E + i];
  } else {
    s = d = i - E;
  }
  float e = hs[s] + hd[d];
  e = (e > 0.f) ? e : 0.2f * e;
  ev[i] = e;
  atomicMax(&mkey[d], flip_f(e));
}
__global__ void k_edge_exp(const int* __restrict__ ei, int E, int n,
                           float* __restrict__ ev, const int* __restrict__ mkey,
                           float* __restrict__ ssum) {
  int i = blockIdx.x * 256 + threadIdx.x;
  if (i >= E + n) return;
  int d = (i < E) ? ei[E + i] : (i - E);
  float ex = __expf(ev[i] - unflip_f(mkey[d]));
  ev[i] = ex;
  atomicAdd(&ssum[d], ex);
}
__global__ void k_gat_init(const float* __restrict__ bias, float* __restrict__ out,
                           int rows) {
  int i = blockIdx.x * 256 + threadIdx.x;
  if (i < rows * FDIM) out[i] = bias[i & 255];
}
// wave per (virtual) edge: out[dst] += h[src] * (ex / ssum[dst])
__global__ __launch_bounds__(256) void k_gat_agg(const int* __restrict__ ei, int E, int n,
                                                 const float* __restrict__ h,
                                                 const float* __restrict__ ev,
                                                 const float* __restrict__ ssum,
                                                 float* __restrict__ out) {
  int lane = threadIdx.x & 31;
  int i = blockIdx.x * 8 + (threadIdx.x >> 5);
  if (i >= E + n) return;
  int s, d;
  if (i < E) {
    s = ei[i];
    d = ei[E + i];
  } else {
    s = d = i - E;
  }
  float alpha = ev[i] / ssum[d];
  const float* hp = h + (size_t)s * FDIM;
  float* op = out + (size_t)d * FDIM;
#pragma unroll
  for (int j = 0; j < 8; ++j) {
    int f = lane + 32 * j;
    atomicAdd(&op[f], hp[f] * alpha);
  }
}

// ---------- final: relu(bn(concat([xg,hg],1))) @ wfc + bfc ----------
__global__ __launch_bounds__(256) void k_final(const float* __restrict__ xg,
                                               const float* __restrict__ hg,
                                               const float* __restrict__ mean,
                                               const float* __restrict__ rstd,
                                               const float* __restrict__ gamma,
                                               const float* __restrict__ beta,
                                               const float* __restrict__ wfc,
                                               const float* __restrict__ bfc,
                                               float* __restrict__ outp, int rows) {
  int lane = threadIdx.x & 31;
  int r = blockIdx.x * 8 + (threadIdx.x >> 5);
  if (r >= rows) return;
  const float* xr = xg + (size_t)r * FDIM;
  const float* hr = hg + (size_t)r * FDIM;
  float s = 0.f;
#pragma unroll
  for (int j = 0; j < 8; ++j) {
    int c = lane + 32 * j;
    float v = (xr[c] - mean[c]) * rstd[c] * gamma[c] + beta[c];
    s += fmaxf(v, 0.f) * wfc[c];
    int c2 = c + FDIM;
    float u = (hr[c] - mean[c2]) * rstd[c2] * gamma[c2] + beta[c2];
    s += fmaxf(u, 0.f) * wfc[c2];
  }
  for (int off = 16; off > 0; off >>= 1) s += __shfl_xor(s, off, 32);
  if (lane == 0) outp[r] = s + bfc[0];
}

// ---------- host orchestration helpers ----------
static void run_gemm(const float* A, const float* W, float* C, int M, half_t* Ah,
                     half_t* Wh, hipStream_t stream) {
  k_prep_a<<<(M * 16 + 255) / 256, 256, 0, stream>>>(A, Ah, M);
  k_prep_w<<<16, 256, 0, stream>>>(W, Wh);
  k_gemm_wmma<<<((M / 16) * 4 + 7) / 8, 256, 0, stream>>>(Ah, Wh, C, M);
}

static void run_gcn(const float* x, const float* W, const float* bias, const int* edges,
                    int E, int rows, const float* gamma, const float* beta, float* hidden,
                    float* outH, float* deg, float* csum, float* csq, float* mean,
                    float* rstd, half_t* Ah, half_t* Wh, hipStream_t stream) {
  run_gemm(x, W, hidden, rows, Ah, Wh, stream);
  k_fill_f<<<(rows + 255) / 256, 256, 0, stream>>>(deg, 1.0f, rows);  // self loop
  k_degree<<<(E + 255) / 256, 256, 0, stream>>>(edges + E, E, deg);
  k_rsqrt_ip<<<(rows + 255) / 256, 256, 0, stream>>>(deg, rows);
  k_gcn_init<<<(rows * FDIM + 255) / 256, 256, 0, stream>>>(hidden, deg, bias, outH,
                                                            rows);
  k_gcn_agg<<<(E + 7) / 8, 256, 0, stream>>>(edges, edges + E, E, hidden, deg, outH);
  k_fill_f<<<1, 256, 0, stream>>>(csum, 0.f, 256);
  k_fill_f<<<1, 256, 0, stream>>>(csq, 0.f, 256);
  k_colstats<<<(rows + 63) / 64, 256, 0, stream>>>(outH, rows, csum, csq);
  k_bn_finalize<<<1, 256, 0, stream>>>(csum, csq, rows, 256, mean, rstd);
  k_bn_apply_relu<<<(rows * FDIM + 255) / 256, 256, 0, stream>>>(outH, mean, rstd, gamma,
                                                                 beta, rows);
}

static void run_gat(const float* xin, const float* W, const float* asrc,
                    const float* adst, const float* bias, const int* ei, float* hidden,
                    float* outX, float* hs, float* hd, int* mkey, float* ssum, float* ev,
                    half_t* Ah, half_t* Wh, int relu, hipStream_t stream) {
  run_gemm(xin, W, hidden, NN, Ah, Wh, stream);
  k_rowdot2<<<(NN + 7) / 8, 256, 0, stream>>>(hidden, asrc, adst, hs, hd, NN);
  k_fill_i<<<(NN + 255) / 256, 256, 0, stream>>>(mkey, (int)0x80000000, NN);
  k_fill_f<<<(NN + 255) / 256, 256, 0, stream>>>(ssum, 0.f, NN);
  k_edge_logits<<<(EEC + 255) / 256, 256, 0, stream>>>(ei, EDC, NN, hs, hd, ev, mkey);
  k_edge_exp<<<(EEC + 255) / 256, 256, 0, stream>>>(ei, EDC, NN, ev, mkey, ssum);
  k_gat_init<<<(NN * FDIM + 255) / 256, 256, 0, stream>>>(bias, outX, NN);  // xin dead
  k_gat_agg<<<(EEC + 7) / 8, 256, 0, stream>>>(ei, EDC, NN, hidden, ev, ssum, outX);
  if (relu) k_relu_ip<<<(NN * FDIM + 255) / 256, 256, 0, stream>>>(outX, NN * FDIM);
}

extern "C" void kernel_launch(void* const* d_in, const int* in_sizes, int n_in,
                              void* d_out, int out_size, void* d_ws, size_t ws_size,
                              hipStream_t stream) {
  (void)in_sizes; (void)n_in; (void)out_size; (void)ws_size;
  const float* x_ab = (const float*)d_in[0];
  const float* x_ag = (const float*)d_in[1];
  const float* W_gcn = (const float*)d_in[2];
  const float* b_gcn = (const float*)d_in[3];
  const float* g1 = (const float*)d_in[4];
  const float* be1 = (const float*)d_in[5];
  const float* W_aggcn = (const float*)d_in[6];
  const float* b_aggcn = (const float*)d_in[7];
  const float* ag_g1 = (const float*)d_in[8];
  const float* ag_be1 = (const float*)d_in[9];
  const float* W_gat = (const float*)d_in[10];
  const float* a_src = (const float*)d_in[11];
  const float* a_dst = (const float*)d_in[12];
  const float* b_gat = (const float*)d_in[13];
  const float* W_gat2 = (const float*)d_in[14];
  const float* a_src2 = (const float*)d_in[15];
  const float* a_dst2 = (const float*)d_in[16];
  const float* b_gat2 = (const float*)d_in[17];
  const float* ag_g2 = (const float*)d_in[18];
  const float* ag_be2 = (const float*)d_in[19];
  const float* W_agfc = (const float*)d_in[20];
  const float* b_agfc = (const float*)d_in[21];
  const float* g2 = (const float*)d_in[22];
  const float* be2 = (const float*)d_in[23];
  const float* W_fc = (const float*)d_in[24];
  const float* b_fc = (const float*)d_in[25];
  const int* e_ab = (const int*)d_in[26];
  const int* e_ag = (const int*)d_in[27];
  const int* e_d = (const int*)d_in[28];
  float* out = (float*)d_out;

  // workspace layout (floats): H | T0 | T1 | deg | hs | hd | mkey | ssum | ev | stats | Ah | Wh
  float* H = (float*)d_ws;              // NN*F : GCN outputs (ab rows 0..N1, ag after)
  float* T0 = H + (size_t)NN * FDIM;    // NN*F : GEMM hidden
  float* T1 = T0 + (size_t)NN * FDIM;   // NN*F : GAT output (in-place ping for GAT2)
  float* deg = T1 + (size_t)NN * FDIM;  // NN
  float* hs = deg + NN;                 // NN
  float* hd = hs + NN;                  // NN
  int* mkey = (int*)(hd + NN);          // NN
  float* ssum = (float*)(mkey + NN);    // NN
  float* ev = ssum + NN;                // EEC
  float* csum = ev + EEC;               // 512
  float* csq = csum + 512;              // 512
  float* mean = csq + 512;              // 512
  float* rstd = mean + 512;             // 512
  half_t* Ah = (half_t*)(rstd + 512);   // NN*F halfs (WMMA A fragments)
  half_t* Wh = Ah + (size_t)NN * FDIM;  // 256*256 halfs (WMMA B fragments)

  // GCN on each graph -> H (with BN + ReLU in place)
  run_gcn(x_ab, W_gcn, b_gcn, e_ab, E1C, N1, g1, be1, T0, H, deg, csum, csq, mean, rstd,
          Ah, Wh, stream);
  run_gcn(x_ag, W_aggcn, b_aggcn, e_ag, E2C, N2, ag_g1, ag_be1, T0,
          H + (size_t)N1 * FDIM, deg, csum, csq, mean, rstd, Ah, Wh, stream);

  // GAT layer 1: H -> T1 (relu), GAT layer 2: T1 -> T1
  run_gat(H, W_gat, a_src, a_dst, b_gat, e_d, T0, T1, hs, hd, mkey, ssum, ev, Ah, Wh, 1,
          stream);
  run_gat(T1, W_gat2, a_src2, a_dst2, b_gat2, e_d, T0, T1, hs, hd, mkey, ssum, ev, Ah,
          Wh, 0, stream);

  // Final AB head: relu(bn(concat([x1, h_ab],1))) @ W_fc + b_fc
  k_fill_f<<<2, 256, 0, stream>>>(csum, 0.f, 512);
  k_fill_f<<<2, 256, 0, stream>>>(csq, 0.f, 512);
  k_colstats2<<<(N1 + 63) / 64, 512, 0, stream>>>(T1, H, N1, csum, csq);
  k_bn_finalize<<<2, 256, 0, stream>>>(csum, csq, N1, 512, mean, rstd);
  k_final<<<(N1 + 7) / 8, 256, 0, stream>>>(T1, H, mean, rstd, g2, be2, W_fc, b_fc, out,
                                            N1);

  // Final AG head: relu(bn(concat([x2, h_ag],1))) @ W_agfc + b_agfc
  k_fill_f<<<2, 256, 0, stream>>>(csum, 0.f, 512);
  k_fill_f<<<2, 256, 0, stream>>>(csq, 0.f, 512);
  k_colstats2<<<(N2 + 63) / 64, 512, 0, stream>>>(T1 + (size_t)N1 * FDIM,
                                                  H + (size_t)N1 * FDIM, N2, csum, csq);
  k_bn_finalize<<<2, 256, 0, stream>>>(csum, csq, N2, 512, mean, rstd);
  k_final<<<(N2 + 7) / 8, 256, 0, stream>>>(T1 + (size_t)N1 * FDIM,
                                            H + (size_t)N1 * FDIM, mean, rstd, ag_g2,
                                            ag_be2, W_agfc, b_agfc, out + N1, N2);
}
